// Decoder_52853867544862
// MI455X (gfx1250) — compile-verified, matrix operand order checked by reading, not run on hardware
//
#include <hip/hip_runtime.h>
#include <math.h>

typedef __attribute__((ext_vector_type(16))) __bf16 v16bf;
typedef __attribute__((ext_vector_type(8)))  float  v8f;

#define B_   64
#define TIN  64
#define TDEC 47
#define VV   32000
#define EE   512
#define HH   1024
#define KZ   2560            // E + H + H  (concat LSTM input length)

__device__ __forceinline__ __bf16 f2bf(float f) {
  unsigned u = __builtin_bit_cast(unsigned, f);
  unsigned r = (u + 0x7FFFu + ((u >> 16) & 1u)) >> 16;
  unsigned short s = (unsigned short)r;
  return __builtin_bit_cast(__bf16, s);
}

// Per-lane K index for element e (0..15) of a 16-bit WMMA A/B fragment,
// per CDNA5 ISA 7.12.2: VGPR0..3 hold K = half*8 + {0..7}, VGPR4..7 hold
// K = 16 + half*8 + {0..7}.
__device__ __forceinline__ int kmap(int e, int half) {
  return ((e < 8) ? e : (e + 8)) + half * 8;
}

union FragU { uint4 q[2]; v16bf v; };

// ---------------------------------------------------------------------------
// Pack a row-major fp32 weight [K x N] (optionally two stacked sources split
// at row `split`) into bf16 WMMA B-fragment tiles: tile (kt,nt) holds
// 32 lanes x 16 elements contiguously so GEMM-time B loads are 2x b128/lane.
// ---------------------------------------------------------------------------
__global__ __launch_bounds__(256) void pack_w(const float* __restrict__ W0,
                                              const float* __restrict__ W1,
                                              int split, int K, int N,
                                              __bf16* __restrict__ out) {
  size_t i = (size_t)blockIdx.x * 256 + threadIdx.x;
  size_t total = (size_t)K * N;
  if (i >= total) return;
  int e    = (int)(i & 15);
  int lane = (int)((i >> 4) & 31);
  size_t tile = i >> 9;
  int Ntiles = N >> 4;
  int nt = (int)(tile % Ntiles);
  int kt = (int)(tile / Ntiles);
  int k = kt * 32 + kmap(e, lane >> 4);
  int n = nt * 16 + (lane & 15);
  float v = (k < split) ? W0[(size_t)k * N + n]
                        : W1[(size_t)(k - split) * N + n];
  out[i] = f2bf(v);
}

__global__ __launch_bounds__(256) void f32_to_bf16(const float* __restrict__ x,
                                                   __bf16* __restrict__ y, int n) {
  int i = blockIdx.x * 256 + threadIdx.x;
  if (i < n) y[i] = f2bf(x[i]);
}

// Az[b][0:512] = bf16(emb[tokens[b][t]])  (teacher-forced input for step t)
__global__ __launch_bounds__(256) void gather_x(const int* __restrict__ tokens,
                                                const float* __restrict__ emb,
                                                int t, __bf16* __restrict__ Az) {
  int i = blockIdx.x * 256 + threadIdx.x;
  if (i >= B_ * EE) return;
  int b = i >> 9, e = i & (EE - 1);
  int tok = tokens[b * TDEC + t];
  Az[(size_t)b * KZ + e] = f2bf(emb[(size_t)tok * EE + e]);
}

__global__ __launch_bounds__(256) void init_state(const float* __restrict__ enc_h,
                                                  const float* __restrict__ enc_c,
                                                  float* __restrict__ c,
                                                  __bf16* __restrict__ Az) {
  int i = blockIdx.x * 256 + threadIdx.x;
  if (i >= B_ * HH) return;
  int b = i >> 10, j = i & (HH - 1);
  c[i] = enc_c[i];
  Az[(size_t)b * KZ + EE + HH + j] = f2bf(enc_h[i]);                    // h
  Az[(size_t)b * KZ + EE + j] = __builtin_bit_cast(__bf16, (unsigned short)0); // attn=0
}

// --- fragment helpers: constant indices only, so everything stays in VGPRs ---
__device__ __forceinline__ void load_a(FragU& f, const __bf16* arow, int kt) {
  const __bf16* p = arow + kt * 32;
  f.q[0] = *(const uint4*)(p);
  f.q[1] = *(const uint4*)(p + 16);
}
__device__ __forceinline__ void load_b4(FragU f[4], const __bf16* bb0,
                                        size_t bstride, int kt) {
  const __bf16* p = bb0 + (size_t)kt * bstride;
#pragma unroll
  for (int j = 0; j < 4; ++j) {
    f[j].q[0] = *(const uint4*)(p + j * 512);
    f[j].q[1] = *(const uint4*)(p + j * 512 + 8);
  }
}
__device__ __forceinline__ void mma4(v8f acc[4], const FragU& a, const FragU b[4]) {
#pragma unroll
  for (int j = 0; j < 4; ++j)
    acc[j] = __builtin_amdgcn_wmma_f32_16x16x32_bf16(
        false, a.v, false, b[j].v, (short)0, acc[j], false, false);
}

// ---------------------------------------------------------------------------
// WMMA GEMM: C[M x N] = A[M x K] * B[K x N], bf16 in, f32 accumulate.
// A row-major bf16 with stride lda; B pre-swizzled (pack_w).
// 256 threads = 8 waves; wave w computes a 16(M) x 64(N) strip:
//   M = blockIdx.y*64 + (w&3)*16, N16-tiles nt0..nt0+3, nt0 = blockIdx.x*8+(w>>2)*4.
// Software pipeline: manual 2x unroll with two named buffer sets (static
// indices -> no scratch spills); Ktiles is always even here (32/64/80).
// Output: f32 (+optional bias) to Cf, or bf16 to Cb.
// ---------------------------------------------------------------------------
__global__ __launch_bounds__(256) void wmma_gemm(
    const __bf16* __restrict__ A, int lda,
    const __bf16* __restrict__ Bp, int Ktiles, int NtilesTot,
    float* __restrict__ Cf, __bf16* __restrict__ Cb, size_t ldc,
    const float* __restrict__ bias) {
  int tid  = threadIdx.x;
  int w    = tid >> 5;
  int lane = tid & 31;
  int half = lane >> 4;
  int mr   = lane & 15;
  int mbase = blockIdx.y * 64 + (w & 3) * 16;
  int nt0   = blockIdx.x * 8 + (w >> 2) * 4;

  const __bf16* arow = A + (size_t)(mbase + mr) * lda + half * 8;
  const __bf16* bb0  = Bp + ((size_t)nt0 << 9) + lane * 16;
  size_t bstride = (size_t)NtilesTot << 9;   // elements per K-tile row of packed B

  v8f acc[4] = {{}, {}, {}, {}};
  FragU a0, a1, b0[4], b1[4];

  load_a(a0, arow, 0);
  load_b4(b0, bb0, bstride, 0);

  for (int kt = 0; kt < Ktiles; kt += 2) {
    // stage kt+1 while computing kt
    load_a(a1, arow, kt + 1);
    load_b4(b1, bb0, bstride, kt + 1);
    int k2 = (kt + 2 < Ktiles) ? (kt + 2) : kt;   // clamp: uniform, branch-free
    __builtin_prefetch((const void*)(bb0 + (size_t)k2 * bstride), 0, 1);
    mma4(acc, a0, b0);
    // stage kt+2 while computing kt+1
    load_a(a0, arow, k2);
    load_b4(b0, bb0, bstride, k2);
    mma4(acc, a1, b1);
  }

  // D layout: VGPR r -> M = mbase + r + 8*half, N = nt*16 + (lane&15)
  if (Cb) {
#pragma unroll
    for (int j = 0; j < 4; ++j) {
      int col = (nt0 + j) * 16 + mr;
      for (int r = 0; r < 8; ++r)
        Cb[(size_t)(mbase + r + 8 * half) * ldc + col] = f2bf(acc[j][r]);
    }
  } else {
#pragma unroll
    for (int j = 0; j < 4; ++j) {
      int col = (nt0 + j) * 16 + mr;
      float bv = bias ? bias[col] : 0.0f;
      for (int r = 0; r < 8; ++r)
        Cf[(size_t)(mbase + r + 8 * half) * ldc + col] = acc[j][r] + bv;
    }
  }
}

// Keras gate order i,f,g,o.  c2 = sig(f)*c + sig(i)*tanh(g); h2 = sig(o)*tanh(c2)
// Writes h2 (f32 for attention), bf16 h2 into Az[:,1536:] and hc[:,0:1024].
__global__ __launch_bounds__(256) void lstm_gates(
    const float* __restrict__ z, const float* __restrict__ bias,
    float* __restrict__ c, float* __restrict__ h2f,
    __bf16* __restrict__ Az, __bf16* __restrict__ hc_bf) {
  int i = blockIdx.x * 256 + threadIdx.x;
  if (i >= B_ * HH) return;
  int b = i >> 10, j = i & (HH - 1);
  const float* zr = z + (size_t)b * 4 * HH;
  float zi = zr[j]          + bias[j];
  float zf = zr[HH + j]     + bias[HH + j];
  float zg = zr[2 * HH + j] + bias[2 * HH + j];
  float zo = zr[3 * HH + j] + bias[3 * HH + j];
  float si = 1.f / (1.f + __expf(-zi));
  float sf = 1.f / (1.f + __expf(-zf));
  float so = 1.f / (1.f + __expf(-zo));
  float cn = sf * c[i] + si * tanhf(zg);
  float hn = so * tanhf(cn);
  c[i]   = cn;
  h2f[i] = hn;
  __bf16 hb = f2bf(hn);
  Az[(size_t)b * KZ + EE + HH + j] = hb;      // h input for next step's z GEMM
  hc_bf[(size_t)b * 2 * HH + j] = hb;          // left half of [h2 | ctx]
}

// One workgroup per batch row: score = h2.keys, softmax, ctx = align.memory
__global__ __launch_bounds__(256) void attention(
    const float* __restrict__ h2f, const float* __restrict__ keys,
    const float* __restrict__ memory, __bf16* __restrict__ hc_bf) {
  __shared__ float sh[HH];
  __shared__ float red[256];
  __shared__ float sc[TIN];
  __shared__ float al[TIN];
  int b = blockIdx.x, tid = threadIdx.x;
  for (int j = tid; j < HH; j += 256) sh[j] = h2f[(size_t)b * HH + j];
  __syncthreads();
  int t = tid & 63, part = tid >> 6;                    // 4 partial sums per t
  const float* krow = keys + ((size_t)b * TIN + t) * HH + part * 256;
  float s = 0.f;
  for (int j = 0; j < 256; ++j) s += sh[part * 256 + j] * krow[j];
  red[tid] = s;
  __syncthreads();
  if (tid < TIN) sc[tid] = red[tid] + red[tid + 64] + red[tid + 128] + red[tid + 192];
  __syncthreads();
  float mx = -3.4e38f;
  for (int u = 0; u < TIN; ++u) mx = fmaxf(mx, sc[u]);
  if (tid < TIN) al[tid] = __expf(sc[tid] - mx);
  __syncthreads();
  float sum = 0.f;
  for (int u = 0; u < TIN; ++u) sum += al[u];
  float inv = 1.f / sum;
  for (int j = tid; j < HH; j += 256) {
    const float* mcol = memory + (size_t)b * TIN * HH + j;
    float acc = 0.f;
    for (int u = 0; u < TIN; ++u) acc += al[u] * mcol[(size_t)u * HH];
    hc_bf[(size_t)b * 2 * HH + HH + j] = f2bf(acc * inv); // right half of [h2 | ctx]
  }
}

extern "C" void kernel_launch(void* const* d_in, const int* in_sizes, int n_in,
                              void* d_out, int out_size, void* d_ws, size_t ws_size,
                              hipStream_t stream) {
  const int*   tokens = (const int*)  d_in[0];
  const float* memory = (const float*)d_in[1];
  const float* enc_h  = (const float*)d_in[2];
  const float* enc_c  = (const float*)d_in[3];
  const float* emb    = (const float*)d_in[4];
  const float* Wm     = (const float*)d_in[5];
  const float* Wa     = (const float*)d_in[6];
  const float* lstm_k = (const float*)d_in[7];
  const float* lstm_r = (const float*)d_in[8];
  const float* lstm_b = (const float*)d_in[9];
  const float* fc_w   = (const float*)d_in[10];
  const float* fc_b   = (const float*)d_in[11];
  float* out = (float*)d_out;
  (void)in_sizes; (void)n_in; (void)out_size; (void)ws_size;

  char* p = (char*)d_ws;
  auto carve = [&](size_t bytes) {
    void* r = (void*)p; p += (bytes + 255) & ~(size_t)255; return r;
  };
  __bf16* Wz_p   = (__bf16*)carve((size_t)KZ * 4096 * 2);     // [lstm_k; lstm_r] packed
  __bf16* Wa_p   = (__bf16*)carve((size_t)2048 * 1024 * 2);
  __bf16* Wm_p   = (__bf16*)carve((size_t)1024 * 1024 * 2);
  __bf16* Wfc_p  = (__bf16*)carve((size_t)1024 * (size_t)VV * 2);
  __bf16* mem_bf = (__bf16*)carve((size_t)B_ * TIN * HH * 2);
  float*  keys   = (float*) carve((size_t)B_ * TIN * HH * 4);
  float*  zbuf   = (float*) carve((size_t)B_ * 4 * HH * 4);
  float*  cbuf   = (float*) carve((size_t)B_ * HH * 4);
  float*  h2f    = (float*) carve((size_t)B_ * HH * 4);
  __bf16* hc_bf  = (__bf16*)carve((size_t)B_ * 2 * HH * 2);
  __bf16* Az     = (__bf16*)carve((size_t)B_ * KZ * 2);       // [x_t | attn | h] bf16

  // --- one-time pre-pass (re-run every call: deterministic) ---
  { size_t n = (size_t)KZ * 4096;
    pack_w<<<(n + 255) / 256, 256, 0, stream>>>(lstm_k, lstm_r, 1536, KZ, 4096, Wz_p); }
  { size_t n = (size_t)2048 * 1024;
    pack_w<<<(n + 255) / 256, 256, 0, stream>>>(Wa, Wa, 2048, 2048, 1024, Wa_p); }
  { size_t n = (size_t)1024 * 1024;
    pack_w<<<(n + 255) / 256, 256, 0, stream>>>(Wm, Wm, 1024, 1024, 1024, Wm_p); }
  { size_t n = (size_t)1024 * VV;
    pack_w<<<(n + 255) / 256, 256, 0, stream>>>(fc_w, fc_w, 1024, 1024, VV, Wfc_p); }
  { int n = B_ * TIN * HH;
    f32_to_bf16<<<(n + 255) / 256, 256, 0, stream>>>(memory, mem_bf, n); }
  { int n = B_ * HH;
    init_state<<<(n + 255) / 256, 256, 0, stream>>>(enc_h, enc_c, cbuf, Az); }

  // keys_att = memory @ Wm   (M = B*TIN = 4096, K = N = 1024)
  wmma_gemm<<<dim3(HH / 128, (B_ * TIN) / 64), 256, 0, stream>>>(
      mem_bf, HH, Wm_p, HH / 32, HH / 16, keys, nullptr, (size_t)HH, nullptr);

  // --- sequential decode: stream ordering carries the recurrence ---
  for (int t = 0; t < TDEC; ++t) {
    gather_x<<<(B_ * EE + 255) / 256, 256, 0, stream>>>(tokens, emb, t, Az);
    // z = Az @ [lstm_k; lstm_r]  ->  [64, 4096] f32
    wmma_gemm<<<dim3((4 * HH) / 128, 1), 256, 0, stream>>>(
        Az, KZ, Wz_p, KZ / 32, (4 * HH) / 16, zbuf, nullptr, (size_t)(4 * HH), nullptr);
    lstm_gates<<<(B_ * HH + 255) / 256, 256, 0, stream>>>(
        zbuf, lstm_b, cbuf, h2f, Az, hc_bf);
    attention<<<B_, 256, 0, stream>>>(h2f, keys, memory, hc_bf);
    // attn2 = [h2 | ctx] @ Wa  -> bf16 straight into Az[:,512:1536]
    wmma_gemm<<<dim3(HH / 128, 1), 256, 0, stream>>>(
        hc_bf, 2 * HH, Wa_p, (2 * HH) / 32, HH / 16,
        nullptr, Az + EE, (size_t)KZ, nullptr);
    // logits = attn2 @ fc_w + fc_b  ->  out[b][t][v]   (A read from Az[:,512:])
    wmma_gemm<<<dim3(VV / 128, 1), 256, 0, stream>>>(
        Az + EE, KZ, Wfc_p, HH / 32, VV / 16,
        out + (size_t)t * VV, nullptr, (size_t)TDEC * VV, fc_b);
  }
}
